// Encoder_3874060501076
// MI455X (gfx1250) — compile-verified, matrix operand order checked by reading, not run on hardware
//
#include <hip/hip_runtime.h>
#include <cstdint>
#include <cstddef>

// Problem constants (match reference)
#define T_STEPS 256
#define BATCH   64
#define E_DIM   512
#define H_DIM   1024
#define G_DIM   4096            // 4*H
#define KCAT    1536            // E + H (concatenated GEMM K)
#define SLOT_HALFS (BATCH * KCAT)   // 98304 halfs per timestep operand slot
#define BH      (BATCH * H_DIM)     // 65536

typedef __attribute__((ext_vector_type(16))) _Float16 v16h;
typedef __attribute__((ext_vector_type(8)))  float    v8f;

union Frag16 { v16h h; uint4 u[2]; };

// ---------------------------------------------------------------------------
// Kernel 1: embedding gather + f32->f16 convert into Acat[t][b][0:512]
// Acat layout: [T+1][BATCH][KCAT] halfs; cols 0..511 = e_t, cols 512..1535 = h_{t-1}
// ---------------------------------------------------------------------------
__global__ __launch_bounds__(256) void gather_embed_f16(
    const int* __restrict__ x_seq,          // [T, B]
    const float* __restrict__ embedding,    // [VOCAB, E]
    _Float16* __restrict__ Acat)
{
    size_t idx = (size_t)blockIdx.x * blockDim.x + threadIdx.x;   // T*B*E threads
    if (idx >= (size_t)T_STEPS * BATCH * E_DIM) return;
    int    k   = (int)(idx & (E_DIM - 1));
    size_t row = idx >> 9;                  // row = t*BATCH + b
    int tok = x_seq[row];
    Acat[row * KCAT + k] = (_Float16)embedding[(size_t)tok * E_DIM + k];
}

// ---------------------------------------------------------------------------
// Kernel 2: weight concat + convert:  Wcat[g][0:512]=W_ih[g], [512:1536]=W_hh[g]
// ---------------------------------------------------------------------------
__global__ __launch_bounds__(256) void concat_weights_f16(
    const float* __restrict__ W_ih,         // [4H, E]
    const float* __restrict__ W_hh,         // [4H, H]
    _Float16* __restrict__ Wcat)            // [4H, KCAT]
{
    size_t idx = (size_t)blockIdx.x * blockDim.x + threadIdx.x;
    if (idx >= (size_t)G_DIM * KCAT) return;
    size_t g = idx / KCAT;
    int    k = (int)(idx - g * KCAT);
    float v = (k < E_DIM) ? W_ih[g * E_DIM + k] : W_hh[g * H_DIM + (k - E_DIM)];
    Wcat[idx] = (_Float16)v;
}

// ---------------------------------------------------------------------------
// Kernel 3: init h0 (f16, inside Acat slot 0) and c0 (f32) to zero
// ---------------------------------------------------------------------------
__global__ __launch_bounds__(256) void init_state(
    _Float16* __restrict__ Acat, float* __restrict__ c_state)
{
    int idx = blockIdx.x * blockDim.x + threadIdx.x;   // BH threads
    if (idx >= BH) return;
    int b = idx >> 10;               // / H_DIM
    int j = idx & (H_DIM - 1);
    Acat[(size_t)b * KCAT + E_DIM + j] = (_Float16)0.0f;
    c_state[idx] = 0.0f;
}

// ---------------------------------------------------------------------------
// Kernel 4: one fused LSTM step.
//   gates[64,4096] = [e_t | h_{t-1}](f16) @ Wcat^T(f16)  + b_ih + b_hh   (WMMA f32 acc)
//   i,f,g,o split along columns; c' = f*c + i*g; h' = o*tanh(c')
// Block = 128 threads = 4 waves; wave w owns gate w's 16x16 tile for this
// (row-tile m0, col-tile j0). Gate tiles meet in LDS for the cell update.
// Grid = (BATCH/16, H_DIM/16) = (4, 64).
// ---------------------------------------------------------------------------
__global__ __launch_bounds__(128) void lstm_step(
    const _Float16* __restrict__ Acat_t,    // [BATCH, KCAT] operand for step t
    const _Float16* __restrict__ Wcat,      // [4H, KCAT]
    const float* __restrict__ b_ih,
    const float* __restrict__ b_hh,
    float* __restrict__ c_state,            // [BATCH, H]
    float* __restrict__ out_t,              // d_out + t*BH
    _Float16* __restrict__ Acat_next,       // step t+1 operand (h' goes to cols 512+)
    float* __restrict__ h_final,            // d_out + T*BH      (only if last)
    float* __restrict__ c_final,            // d_out + T*BH + BH (only if last)
    int last)
{
    const int lane = threadIdx.x & 31;
    const int w    = threadIdx.x >> 5;            // gate index 0..3 (i,f,g,o)
    const int m0   = blockIdx.x << 4;             // batch-row tile base
    const int j0   = blockIdx.y << 4;             // hidden-col tile base
    const int nl   = lane & 15;                   // fragment row/col within tile
    const int khalf = (lane < 16) ? 0 : 8;        // ISA f16 A/B K-packing split

    // A: row (m0+nl) of [e_t | h_{t-1}]; B: row (w*H + j0 + nl) of Wcat.
    const _Float16* rowA = Acat_t + (size_t)(m0 + nl) * KCAT;
    const _Float16* rowB = Wcat   + (size_t)(w * H_DIM + j0 + nl) * KCAT;

    // Bias depends only on the gate column this lane accumulates.
    const int gcol = w * H_DIM + j0 + nl;
    const float biasv = b_ih[gcol] + b_hh[gcol];

    v8f acc;
#pragma unroll
    for (int r = 0; r < 8; ++r) acc[r] = biasv;

    // K loop over the concatenated 1536 reduction dim, 32 per WMMA.
#pragma unroll 4
    for (int kb = 0; kb < KCAT; kb += 32) {
        Frag16 a, b;
        // lanes<16: K {kb..kb+7, kb+16..kb+23}; lanes>=16: {kb+8..15, kb+24..31}
        a.u[0] = *reinterpret_cast<const uint4*>(rowA + kb + khalf);
        a.u[1] = *reinterpret_cast<const uint4*>(rowA + kb + 16 + khalf);
        b.u[0] = *reinterpret_cast<const uint4*>(rowB + kb + khalf);
        b.u[1] = *reinterpret_cast<const uint4*>(rowB + kb + 16 + khalf);
        acc = __builtin_amdgcn_wmma_f32_16x16x32_f16(
            /*neg_a=*/false, a.h, /*neg_b=*/false, b.h,
            /*c_mod=*/(short)0, acc, /*reuse_a=*/false, /*reuse_b=*/false);
    }

    // Deposit this wave's gate tile into LDS (C/D layout: lane = col, vgpr = row).
    __shared__ float lds[4][16][17];   // +1 pad: conflict-free row/col access
    {
        const int mhi = (lane >= 16) ? 8 : 0;
#pragma unroll
        for (int r = 0; r < 8; ++r)
            lds[w][r + mhi][nl] = acc[r];
    }
    __syncthreads();

    // Cell update: 256 elements, 128 threads -> 2 each.
    for (int e = threadIdx.x; e < 256; e += 128) {
        const int m = e >> 4;
        const int n = e & 15;
        float iv = lds[0][m][n];
        float fv = lds[1][m][n];
        float gv = lds[2][m][n];
        float ov = lds[3][m][n];
        iv = 1.0f / (1.0f + __expf(-iv));
        fv = 1.0f / (1.0f + __expf(-fv));
        ov = 1.0f / (1.0f + __expf(-ov));
        gv = tanhf(gv);

        const int bb = m0 + m;
        const int j  = j0 + n;
        const size_t sidx = (size_t)bb * H_DIM + j;

        const float cp = c_state[sidx];
        const float cn = fv * cp + iv * gv;
        const float hn = ov * tanhf(cn);

        c_state[sidx] = cn;
        out_t[sidx]   = hn;                                       // outs[t]
        Acat_next[(size_t)bb * KCAT + E_DIM + j] = (_Float16)hn;  // feeds step t+1
        if (last) {
            h_final[sidx] = hn;
            c_final[sidx] = cn;
        }
    }
}

// ---------------------------------------------------------------------------
// Host launcher
// ---------------------------------------------------------------------------
extern "C" void kernel_launch(void* const* d_in, const int* in_sizes, int n_in,
                              void* d_out, int out_size, void* d_ws, size_t ws_size,
                              hipStream_t stream)
{
    const int*   x_seq = (const int*)  d_in[0];   // [T, B] int32
    const float* emb   = (const float*)d_in[1];   // [VOCAB, E]
    const float* W_ih  = (const float*)d_in[2];   // [4H, E]
    const float* W_hh  = (const float*)d_in[3];   // [4H, H]
    const float* b_ih  = (const float*)d_in[4];   // [4H]
    const float* b_hh  = (const float*)d_in[5];   // [4H]
    float* out = (float*)d_out;                   // [T,B,H] ++ [B,H] ++ [B,H]

    // Workspace carve-up (all 256B-aligned chunks):
    //   Wcat : [4H][KCAT]  f16  = 12,582,912 B
    //   Acat : [T+1][B][KCAT] f16 = 50,528,256 B
    //   c    : [B][H] f32        =    262,144 B
    char* ws = (char*)d_ws;
    _Float16* Wcat    = (_Float16*)ws;
    _Float16* Acat    = (_Float16*)(ws + (size_t)12582912);
    float*    c_state = (float*)   (ws + (size_t)12582912 + (size_t)50528256);

    // 1) Embedding gather -> f16 operand slots
    {
        size_t n = (size_t)T_STEPS * BATCH * E_DIM;      // 8,388,608
        gather_embed_f16<<<(unsigned)((n + 255) / 256), 256, 0, stream>>>(x_seq, emb, Acat);
    }
    // 2) Weight concat -> f16 (L2-resident for the whole recurrence)
    {
        size_t n = (size_t)G_DIM * KCAT;                 // 6,291,456
        concat_weights_f16<<<(unsigned)((n + 255) / 256), 256, 0, stream>>>(W_ih, W_hh, Wcat);
    }
    // 3) Zero h0 (f16, in Acat slot 0) and c0
    init_state<<<(BH + 255) / 256, 256, 0, stream>>>(Acat, c_state);

    // 4) 256 fused WMMA steps (stream order = recurrence order)
    float* h_final = out + (size_t)T_STEPS * BH;
    float* c_final = h_final + BH;
    dim3 grid(BATCH / 16, H_DIM / 16);   // (4, 64)
    for (int t = 0; t < T_STEPS; ++t) {
        lstm_step<<<grid, 128, 0, stream>>>(
            Acat + (size_t)t * SLOT_HALFS,
            Wcat, b_ih, b_hh, c_state,
            out + (size_t)t * BH,
            Acat + (size_t)(t + 1) * SLOT_HALFS,
            h_final, c_final,
            (t == T_STEPS - 1) ? 1 : 0);
    }
}
